// MemoryEfficientMHA_40407052321356
// MI455X (gfx1250) — matrix operand order repaired and checked by reading, not compiled, not run on hardware
//
#include <hip/hip_runtime.h>
#include <hip/hip_bf16.h>

typedef __attribute__((ext_vector_type(16))) _Float16 v16h;
typedef __attribute__((ext_vector_type(8)))  _Float16 v8h;
typedef __attribute__((ext_vector_type(8)))  float    v8f;

#define T_ 4
#define E_ 1024
#define H_ 16
#define D_ 128
#define C_ 2048

// Convert two runs of 8 contiguous f32 into one v16h operand fragment.
__device__ __forceinline__ v16h cvt16(const float* __restrict__ p,
                                      const float* __restrict__ q) {
  const float4* P = (const float4*)p;
  const float4* Q = (const float4*)q;
  float4 x0 = P[0], x1 = P[1], y0 = Q[0], y1 = Q[1];
  v16h r;
  r[0]=(_Float16)x0.x;  r[1]=(_Float16)x0.y;  r[2]=(_Float16)x0.z;  r[3]=(_Float16)x0.w;
  r[4]=(_Float16)x1.x;  r[5]=(_Float16)x1.y;  r[6]=(_Float16)x1.z;  r[7]=(_Float16)x1.w;
  r[8]=(_Float16)y0.x;  r[9]=(_Float16)y0.y;  r[10]=(_Float16)y0.z; r[11]=(_Float16)y0.w;
  r[12]=(_Float16)y1.x; r[13]=(_Float16)y1.y; r[14]=(_Float16)y1.z; r[15]=(_Float16)y1.w;
  return r;
}

__device__ __forceinline__ v16h join8(v8h a, v8h b) {
  v16h r;
  #pragma unroll
  for (int i = 0; i < 8; i++) { r[i] = a[i]; r[i+8] = b[i]; }
  return r;
}

__device__ __forceinline__ v8f wmma_f16(v16h a, v16h b, v8f c) {
  return __builtin_amdgcn_wmma_f32_16x16x32_f16(false, a, false, b,
                                                (short)0, c, false, false);
}

// ---------------------------------------------------------------------------
// Stage 1: QKV projections.  out(m,n) = sum_k in[m][k]*W[n][k] + b[n]
// A-frag (16x32 f16): lane row = lane&15, K runs {8hi..8hi+7} and {16+8hi..}
// B-frag (32x16 f16): lane col = lane&15, K runs {8hi..8hi+7} and {16+8hi..}
// Q scaled by 1/sqrt(D); V stored transposed [T][H][D][E] for the PV matmul.
// ---------------------------------------------------------------------------
__global__ void __launch_bounds__(256)
qkv_proj_kernel(const float* __restrict__ query,
                const float* __restrict__ key,
                const float* __restrict__ value,
                const float* __restrict__ Wq, const float* __restrict__ bq,
                const float* __restrict__ Wk, const float* __restrict__ bk,
                const float* __restrict__ Wv, const float* __restrict__ bv,
                _Float16* __restrict__ q_ws, _Float16* __restrict__ k_ws,
                _Float16* __restrict__ vt_ws) {
  const int lane = threadIdx.x & 31;
  const int wid  = threadIdx.x >> 5;
  const int lr = lane & 15;
  const int hi = lane >> 4;
  const int proj = blockIdx.z;
  const int m0 = blockIdx.x * 16;
  const int n0 = (blockIdx.y * 8 + wid) * 16;

  const float* in; const float* W; const float* bias;
  if (proj == 0)      { in = query; W = Wq; bias = bq; }
  else if (proj == 1) { in = key;   W = Wk; bias = bk; }
  else                { in = value; W = Wv; bias = bv; }

  const int m = m0 + lr;
  const int n = n0 + lr;
  const float* arow = in + (size_t)m * D_;
  const float* brow = W  + (size_t)n * D_;

  v8f acc = {};
  #pragma unroll
  for (int kc = 0; kc < 4; kc++) {
    v16h a = cvt16(arow + kc*32 + 8*hi, arow + kc*32 + 16 + 8*hi);
    v16h b = cvt16(brow + kc*32 + 8*hi, brow + kc*32 + 16 + 8*hi);
    acc = wmma_f16(a, b, acc);
  }
  const float bn = bias[n];
  const float scale = (proj == 0) ? 0.08838834764831845f : 1.0f;

  const int h = n >> 7;     // column n -> (head, d)
  const int d = n & 127;
  #pragma unroll
  for (int r = 0; r < 8; r++) {
    const int mg = m0 + r + 8*hi;           // token index
    const int t = mg >> 10, e = mg & 1023;
    const float v = (acc[r] + bn) * scale;
    if (proj == 2) {
      vt_ws[(((size_t)t*H_ + h)*D_ + d)*E_ + e] = (_Float16)v;
    } else {
      _Float16* dst = (proj == 0) ? q_ws : k_ws;
      dst[(((size_t)t*H_ + h)*E_ + e)*D_ + d] = (_Float16)v;
    }
  }
}

// ---------------------------------------------------------------------------
// Stage 2: causal flash attention.  One wave owns a 16-query tile of one
// (t,h) pair and keeps the 16x128 f32 output tile in registers (8 x v8f).
// ---------------------------------------------------------------------------
__global__ void __launch_bounds__(128)
flash_attn_kernel(const _Float16* __restrict__ q_ws,
                  const _Float16* __restrict__ k_ws,
                  const _Float16* __restrict__ vt_ws,
                  _Float16* __restrict__ att_ws) {
  __shared__ _Float16 ldsP[4][16][32];       // per-wave P staging (C->A layout)
  const int lane = threadIdx.x & 31;
  const int wid  = threadIdx.x >> 5;
  const int lr = lane & 15;
  const int hi = lane >> 4;

  const int th = blockIdx.x;                 // t*H + h
  const int qb = blockIdx.y * 4 + wid;       // 16-row query tile
  const int m0 = qb * 16;

  const _Float16* Qp = q_ws  + (size_t)th * E_ * D_;
  const _Float16* Kp = k_ws  + (size_t)th * E_ * D_;
  const _Float16* Vp = vt_ws + (size_t)th * D_ * E_;   // [D][E]

  // Resident Q A-fragments (4 x K=32 chunks)
  v16h qa[4];
  const _Float16* qrow = Qp + (size_t)(m0 + lr) * D_;
  #pragma unroll
  for (int kc = 0; kc < 4; kc++) {
    v8h a0 = *(const v8h*)(qrow + kc*32 + 8*hi);
    v8h a1 = *(const v8h*)(qrow + kc*32 + 16 + 8*hi);
    qa[kc] = join8(a0, a1);
  }

  v8f o[8] = {};
  float rm[8], rl[8];
  #pragma unroll
  for (int r = 0; r < 8; r++) { rm[r] = -3.0e38f; rl[r] = 0.0f; }

  const int kbMax = (m0 + 15) >> 5;          // causal: stop at diagonal block
  for (int kb = 0; kb <= kbMax; kb++) {
    const int ks = kb * 32;
    // Prefetch next key block's K rows while we compute on this one
    // (uniform condition: no lane divergence, EXEC stays all-1s for WMMA).
    if (kb < kbMax) {
      __builtin_prefetch(Kp + (size_t)(ks + 32 + lr) * D_, 0, 3);
      __builtin_prefetch(Kp + (size_t)(ks + 48 + lr) * D_, 0, 3);
    }
    // ---- S = Q * K^T : two 16x16 score tiles over this 32-key block
    v8f s0 = {}, s1 = {};
    const _Float16* k0row = Kp + (size_t)(ks + lr) * D_;
    const _Float16* k1row = Kp + (size_t)(ks + 16 + lr) * D_;
    #pragma unroll
    for (int kc = 0; kc < 4; kc++) {
      v16h b0 = join8(*(const v8h*)(k0row + kc*32 + 8*hi), *(const v8h*)(k0row + kc*32 + 16 + 8*hi));
      s0 = wmma_f16(qa[kc], b0, s0);
      v16h b1 = join8(*(const v8h*)(k1row + kc*32 + 8*hi), *(const v8h*)(k1row + kc*32 + 16 + 8*hi));
      s1 = wmma_f16(qa[kc], b1, s1);
    }
    // ---- causal mask + online softmax (rows live in 16-lane halves)
    #pragma unroll
    for (int r = 0; r < 8; r++) {
      const int mg = m0 + r + 8*hi;
      if (ks + lr > mg)      s0[r] = -3.0e38f;
      if (ks + 16 + lr > mg) s1[r] = -3.0e38f;
      float v = fmaxf(s0[r], s1[r]);
      v = fmaxf(v, __shfl_xor(v, 1, 32));
      v = fmaxf(v, __shfl_xor(v, 2, 32));
      v = fmaxf(v, __shfl_xor(v, 4, 32));
      v = fmaxf(v, __shfl_xor(v, 8, 32));
      const float nm = fmaxf(rm[r], v);
      const float sc = __expf(rm[r] - nm);
      rm[r] = nm;
      const float p0 = __expf(s0[r] - nm);
      const float p1 = __expf(s1[r] - nm);
      s0[r] = p0; s1[r] = p1;
      float rs = p0 + p1;
      rs += __shfl_xor(rs, 1, 32);
      rs += __shfl_xor(rs, 2, 32);
      rs += __shfl_xor(rs, 4, 32);
      rs += __shfl_xor(rs, 8, 32);
      rl[r] = rl[r] * sc + rs;
      #pragma unroll
      for (int j = 0; j < 8; j++) o[j][r] *= sc;
    }
    // ---- P: C-layout -> A-layout via per-wave LDS bounce
    #pragma unroll
    for (int r = 0; r < 8; r++) {
      ldsP[wid][r + 8*hi][lr]      = (_Float16)s0[r];
      ldsP[wid][r + 8*hi][16 + lr] = (_Float16)s1[r];
    }
    asm volatile("s_wait_dscnt 0x0" ::: "memory");
    v8h pl = *(const v8h*)&ldsP[wid][lr][8*hi];
    v8h ph = *(const v8h*)&ldsP[wid][lr][16 + 8*hi];
    v16h pa = join8(pl, ph);
    // ---- O += P * V : 8 d-tiles, V^T rows are contiguous along keys
    #pragma unroll
    for (int j = 0; j < 8; j++) {
      v16h bv = join8(*(const v8h*)(Vp + (size_t)(j*16 + lr) * E_ + ks + 8*hi), *(const v8h*)(Vp + (size_t)(j*16 + lr) * E_ + ks + 16 + 8*hi));
      o[j] = wmma_f16(pa, bv, o[j]);
    }
    // DS ops are in-order within a wave: next iteration's LDS stores cannot
    // bypass this iteration's loads.
  }
  // ---- epilogue: normalize, store f16 as [T][E][H*D]
  const int t = th >> 4, h = th & 15;
  #pragma unroll
  for (int r = 0; r < 8; r++) {
    const float inv = 1.0f / rl[r];
    const int e = m0 + r + 8*hi;
    _Float16* dst = att_ws + ((size_t)t*E_ + e) * C_ + h*D_;
    #pragma unroll
    for (int j = 0; j < 8; j++)
      dst[j*16 + lr] = (_Float16)(o[j][r] * inv);
  }
}

// ---------------------------------------------------------------------------
// Stage 3: output projection.  out = attended(4096x2048) * Wo^T + bo
// ---------------------------------------------------------------------------
__global__ void __launch_bounds__(256)
out_proj_kernel(const _Float16* __restrict__ att_ws,
                const float* __restrict__ Wo,
                const float* __restrict__ bo,
                float* __restrict__ out) {
  const int lane = threadIdx.x & 31;
  const int wid  = threadIdx.x >> 5;
  const int lr = lane & 15;
  const int hi = lane >> 4;
  const int m0 = blockIdx.x * 16;
  const int n  = wid * 16 + lr;              // 8 waves cover N = 128

  const _Float16* arow = att_ws + (size_t)(m0 + lr) * C_;
  const float*    brow = Wo + (size_t)n * C_;

  v8f acc = {};
  #pragma unroll 4
  for (int kc = 0; kc < 64; kc++) {
    v8h a0 = *(const v8h*)(arow + kc*32 + 8*hi);
    v8h a1 = *(const v8h*)(arow + kc*32 + 16 + 8*hi);
    v16h a = join8(a0, a1);
    v16h b = cvt16(brow + kc*32 + 8*hi, brow + kc*32 + 16 + 8*hi);
    acc = wmma_f16(a, b, acc);
  }
  const float bn = bo[n];
  #pragma unroll
  for (int r = 0; r < 8; r++) {
    const int mg = m0 + r + 8*hi;
    out[(size_t)mg * D_ + n] = acc[r] + bn;
  }
}

extern "C" void kernel_launch(void* const* d_in, const int* in_sizes, int n_in,
                              void* d_out, int out_size, void* d_ws, size_t ws_size,
                              hipStream_t stream) {
  const float* key   = (const float*)d_in[0];
  const float* value = (const float*)d_in[1];
  const float* query = (const float*)d_in[2];
  const float* Wq = (const float*)d_in[3];
  const float* bq = (const float*)d_in[4];
  const float* Wk = (const float*)d_in[5];
  const float* bk = (const float*)d_in[6];
  const float* Wv = (const float*)d_in[7];
  const float* bv = (const float*)d_in[8];
  const float* Wo = (const float*)d_in[9];
  const float* bo = (const float*)d_in[10];
  float* out = (float*)d_out;

  const size_t per = (size_t)T_ * H_ * E_ * D_;   // 8M f16 elems = 16 MB each
  _Float16* q_ws   = (_Float16*)d_ws;
  _Float16* k_ws   = q_ws + per;
  _Float16* vt_ws  = k_ws + per;
  _Float16* att_ws = vt_ws + per;

  dim3 g1(T_*E_/16, C_/(16*8), 3);   // (256, 16, 3), 8 waves/block
  hipLaunchKernelGGL(qkv_proj_kernel, g1, dim3(256), 0, stream,
                     query, key, value, Wq, bq, Wk, bk, Wv, bv,
                     q_ws, k_ws, vt_ws);

  dim3 g2(T_*H_, E_/(16*4));         // (64, 16), 4 waves/block
  hipLaunchKernelGGL(flash_attn_kernel, g2, dim3(128), 0, stream,
                     q_ws, k_ws, vt_ws, att_ws);

  dim3 g3(T_*E_/16);                 // 256 blocks, 8 waves cover N=128
  hipLaunchKernelGGL(out_proj_kernel, g3, dim3(256), 0, stream,
                     att_ws, Wo, bo, out);
}
